// CoreMLOptimizedCosmicFish_72808285602072
// MI455X (gfx1250) — matrix operand order repaired and checked by reading, not compiled, not run on hardware
//
#include <hip/hip_runtime.h>
#include <cmath>

typedef __bf16 bf16;
typedef __attribute__((ext_vector_type(16))) __bf16 v16bf;
typedef __attribute__((ext_vector_type(8)))  float  v8f;

static constexpr int kB = 8, kT = 512, kD = 1024, kH = 16, kKVH = 4, kHD = 64, kL = 4, kV = 32000, kFF = 4096;
static constexpr int kQKV = kH * kHD + 2 * kKVH * kHD; // 1536
static constexpr int kBT = kB * kT;                    // 4096

// ---- CDNA5 async global->LDS copy (ASYNCcnt-tracked, no VGPR staging) -----
__device__ __forceinline__ void async_copy_b128(unsigned ldsOff, const void* gaddr)
{
    asm volatile("global_load_async_to_lds_b128 %0, %1, off"
                 :: "v"(ldsOff), "v"((unsigned long long)(uintptr_t)gaddr)
                 : "memory");
}
#define WAIT_ASYNCCNT(n) asm volatile("s_wait_asynccnt " #n ::: "memory")

// ---------------------------------------------------------------------------
// Batched bf16 WMMA GEMM: C = A[M,K] (bf16,row-major,lda) x Bt[N,K]^T (bf16,ldb)
// + bias + residual, optional GELU, optional bf16 output.
// Block = 256 threads = 8 wave32 waves. BM=64, BK=32, BN in {128,64}.
// Double-buffered LDS fed by global_load_async_to_lds_b128; s_wait_asynccnt
// partial waits keep the next tile in flight while WMMAs consume the current.
// Batch z: A += z*aBatch, Bt += (z/bDiv)*bBatch, C += (z/cDiv)*cs1 + (z%cDiv)*cs2.
// ---------------------------------------------------------------------------
template <int BN>
__global__ __launch_bounds__(256) void gemm_bf16_kernel(
    const bf16* __restrict__ A, long long aBatch, int lda,
    const bf16* __restrict__ Bt, long long bBatch, int bDiv, int ldb,
    const float* __restrict__ bias, const float* __restrict__ residual,
    void* __restrict__ Cv, long long cs1, long long cs2, int cDiv, int ldc,
    int K, int flags)
{
    constexpr int BM = 64;
    constexpr int BK = 32;
    constexpr int LDP = BK + 8;      // 40 bf16 = 80B pitch: 16B aligned, bank-conflict-free
    constexpr int WN = BN / 2;       // per-wave N extent
    constexpr int WNF = WN / 16;     // WMMA frags per wave in N

    __shared__ __align__(16) bf16 lA[2][BM * LDP];
    __shared__ __align__(16) bf16 lB[2][BN * LDP];

    const int tid  = threadIdx.x;
    const int lane = tid & 31;
    const int wid  = tid >> 5;
    const int half = lane >> 4;
    const int l16  = lane & 15;

    const int z      = blockIdx.z;
    const int blockM = blockIdx.y * BM;
    const int blockN = blockIdx.x * BN;

    A  += (long long)z * aBatch;
    Bt += (long long)(z / bDiv) * bBatch;
    const long long cOff = (long long)(z / cDiv) * cs1 + (long long)(z % cDiv) * cs2;

    const int waveM = (wid & 3) * 16;
    const int waveN = (wid >> 2) * WN;

    v8f acc[WNF];
#pragma unroll
    for (int f = 0; f < WNF; ++f) acc[f] = (v8f){0.f, 0.f, 0.f, 0.f, 0.f, 0.f, 0.f, 0.f};

    // per-thread async-copy assignments
    const int ar = tid >> 2;          // A tile: 64 rows x 32 cols, 16B per thread
    const int ac = (tid & 3) * 8;
    const unsigned aLds0 = (unsigned)(uintptr_t)(&lA[0][0]) + (unsigned)((ar * LDP + ac) * 2);
    const unsigned aLds1 = (unsigned)(uintptr_t)(&lA[1][0]) + (unsigned)((ar * LDP + ac) * 2);
    const bf16* aG = A + (long long)(blockM + ar) * lda + ac;

    const int br = (BN == 128) ? (tid >> 1) : (tid >> 2);
    const int bc = (BN == 128) ? ((tid & 1) * 16) : ((tid & 3) * 8);
    const unsigned bLds0 = (unsigned)(uintptr_t)(&lB[0][0]) + (unsigned)((br * LDP + bc) * 2);
    const unsigned bLds1 = (unsigned)(uintptr_t)(&lB[1][0]) + (unsigned)((br * LDP + bc) * 2);
    const bf16* bG = Bt + (long long)(blockN + br) * ldb + bc;

    auto issueTiles = [&](int kk, int buf) {
        async_copy_b128(buf ? aLds1 : aLds0, aG + kk);
        if constexpr (BN == 128) {
            const unsigned d = buf ? bLds1 : bLds0;
            async_copy_b128(d,      bG + kk);
            async_copy_b128(d + 16, bG + kk + 8);
        } else {
            async_copy_b128(buf ? bLds1 : bLds0, bG + kk);
        }
    };

    issueTiles(0, 0);                 // prologue: first tile into buffer 0
    int cur = 0;
    for (int kk = 0; kk < K; kk += BK) {
        const bool more = (kk + BK) < K;
        if (more) issueTiles(kk + BK, cur ^ 1);
        // wait until only the just-issued loads remain in flight -> current
        // buffer's data has landed in LDS (async loads complete in order).
        if (more) {
            if constexpr (BN == 128) WAIT_ASYNCCNT(0x3);
            else                     WAIT_ASYNCCNT(0x2);
        } else {
            WAIT_ASYNCCNT(0x0);
        }
        __syncthreads();

        // A fragment (16x32 bf16): lane l16 = row M; VGPR pair i holds K = 2i + 8*((i>>2)+half)
        union { v16bf v; unsigned u[8]; } af;
        const bf16* aRow = &lA[cur][(waveM + l16) * LDP];
#pragma unroll
        for (int i = 0; i < 8; ++i) {
            const int k = 2 * i + 8 * ((i >> 2) + half);
            af.u[i] = *(const unsigned*)(aRow + k);
        }
#pragma unroll
        for (int f = 0; f < WNF; ++f) {
            // B fragment (32x16): lane l16 = column N, K runs of 8 at 8*half and 16+8*half
            union { v16bf v; uint4 q[2]; } bfv;
            const bf16* bRow = &lB[cur][(waveN + f * 16 + l16) * LDP + 8 * half];
            bfv.q[0] = *(const uint4*)(bRow);
            bfv.q[1] = *(const uint4*)(bRow + 16);
            acc[f] = __builtin_amdgcn_wmma_f32_16x16x32_bf16(
                false, af.v, false, bfv.v, (short)0, acc[f], false, false);
        }
        // all waves done reading `cur` before anyone overwrites it next iter
        __syncthreads();
        cur ^= 1;
    }

    const bool gelu = (flags & 1) != 0;
    const bool obf  = (flags & 2) != 0;
#pragma unroll
    for (int f = 0; f < WNF; ++f) {
        const int gn = blockN + waveN + f * 16 + l16;
        const float bsv = bias ? bias[gn] : 0.0f;
#pragma unroll
        for (int r = 0; r < 8; ++r) {
            const int gm = blockM + waveM + 8 * half + r;  // D layout: row = r + 8*half
            float v = acc[f][r] + bsv;
            if (residual) v += residual[(long long)gm * ldc + gn];
            if (gelu) v = 0.5f * v * (1.0f + erff(v * 0.70710678118654752f));
            const long long ci = cOff + (long long)gm * ldc + gn;
            if (obf) ((bf16*)Cv)[ci] = (bf16)v;
            else     ((float*)Cv)[ci] = v;
        }
    }
}

// ---------------------------------------------------------------------------
__global__ void embed_kernel(const int* __restrict__ ids, const float* __restrict__ wte,
                             float* __restrict__ out)
{
    const int i = blockIdx.x * blockDim.x + threadIdx.x;  // [0, BT*D)
    const int tok = i >> 10;
    const int d   = i & (kD - 1);
    out[i] = wte[(long long)ids[tok] * kD + d];
}

// LayerNorm over D=1024, one 256-thread block per row, bf16 output.
__global__ __launch_bounds__(256) void layernorm_kernel(
    const float* __restrict__ x, const float* __restrict__ g,
    const float* __restrict__ b, bf16* __restrict__ out)
{
    const int row = blockIdx.x;
    const float* xr = x + (long long)row * kD;
    float vals[4], s = 0.f, s2 = 0.f;
#pragma unroll
    for (int i = 0; i < 4; ++i) {
        const float v = xr[threadIdx.x + i * 256];
        vals[i] = v; s += v; s2 += v * v;
    }
    __shared__ float red0[256], red1[256];
    red0[threadIdx.x] = s; red1[threadIdx.x] = s2;
    __syncthreads();
    for (int off = 128; off > 0; off >>= 1) {
        if (threadIdx.x < off) {
            red0[threadIdx.x] += red0[threadIdx.x + off];
            red1[threadIdx.x] += red1[threadIdx.x + off];
        }
        __syncthreads();
    }
    const float mean = red0[0] * (1.0f / kD);
    const float var  = red1[0] * (1.0f / kD) - mean * mean;
    const float rstd = rsqrtf(var + 1e-5f);
#pragma unroll
    for (int i = 0; i < 4; ++i) {
        const int c = threadIdx.x + i * 256;
        out[(long long)row * kD + c] = (bf16)((vals[i] - mean) * rstd * g[c] + b[c]);
    }
}

// fp32 [K,N] -> bf16 [N,K] tiled transpose (weights). block(32,8), grid(N/32,K/32).
__global__ __launch_bounds__(256) void transpose_cvt_kernel(
    const float* __restrict__ in, bf16* __restrict__ out, int Kdim, int Ndim)
{
    __shared__ float tile[32][33];
    const int nx = blockIdx.x * 32 + threadIdx.x;
    const int ky = blockIdx.y * 32;
#pragma unroll
    for (int i = 0; i < 32; i += 8)
        tile[threadIdx.y + i][threadIdx.x] = in[(long long)(ky + threadIdx.y + i) * Ndim + nx];
    __syncthreads();
    const int on = blockIdx.x * 32 + threadIdx.y;
    const int ok = ky + threadIdx.x;
#pragma unroll
    for (int i = 0; i < 32; i += 8)
        out[(long long)(on + i) * Kdim + ok] = (bf16)tile[threadIdx.x][threadIdx.y + i];
}

// RoPE on q, folds 1/sqrt(HD)=0.125 into q. Layout out: [B*H, T, HD] bf16.
__global__ void rope_q_kernel(const float* __restrict__ qkv, bf16* __restrict__ qbf)
{
    const int i = blockIdx.x * blockDim.x + threadIdx.x;  // B*T*H*(HD/2)
    const int p = i & 31, h = (i >> 5) & 15, t = (i >> 9) & 511, b = i >> 18;
    const float* src = qkv + (long long)(b * kT + t) * kQKV + h * kHD + 2 * p;
    const float ue = src[0], uo = src[1];
    const float inv = expf(-(float)(2 * p) * (9.210340371976184f / 64.f)); // 10000^(-2p/64)
    float sn, cs; sincosf((float)t * inv, &sn, &cs);
    bf16* dst = qbf + ((long long)(b * kH + h) * kT + t) * kHD + 2 * p;
    dst[0] = (bf16)((ue * cs - uo * sn) * 0.125f);
    dst[1] = (bf16)((ue * sn + uo * cs) * 0.125f);
}

// RoPE on k. Layout out: [B*KVH, T, HD] bf16 (acts as Bt for the scores GEMM).
__global__ void rope_k_kernel(const float* __restrict__ qkv, bf16* __restrict__ kbf)
{
    const int i = blockIdx.x * blockDim.x + threadIdx.x;  // B*T*KVH*(HD/2)
    const int p = i & 31, kh = (i >> 5) & 3, t = (i >> 7) & 511, b = i >> 16;
    const float* src = qkv + (long long)(b * kT + t) * kQKV + kH * kHD + kh * kHD + 2 * p;
    const float ue = src[0], uo = src[1];
    const float inv = expf(-(float)(2 * p) * (9.210340371976184f / 64.f));
    float sn, cs; sincosf((float)t * inv, &sn, &cs);
    bf16* dst = kbf + ((long long)(b * kKVH + kh) * kT + t) * kHD + 2 * p;
    dst[0] = (bf16)(ue * cs - uo * sn);
    dst[1] = (bf16)(ue * sn + uo * cs);
}

// V transposed to [B*KVH, HD, T] bf16 (acts as Bt for the attn*V GEMM). Coalesced writes.
__global__ void vtrans_kernel(const float* __restrict__ qkv, bf16* __restrict__ vt)
{
    const int i = blockIdx.x * blockDim.x + threadIdx.x;  // B*KVH*HD*T
    const int t = i & 511, d = (i >> 9) & 63, kh = (i >> 15) & 3, b = i >> 17;
    vt[i] = (bf16)qkv[(long long)(b * kT + t) * kQKV + (kH + kKVH) * kHD + kh * kHD + d];
}

// Causal softmax over rows of S [B*H, T, T] fp32; writes bf16 probs IN PLACE
// (reinterpreting the same buffer) -- all fp32 reads complete before the
// reduction barriers that gate the bf16 writes.
__global__ __launch_bounds__(256) void softmax_causal_kernel(float* __restrict__ S)
{
    const int m = blockIdx.x, z = blockIdx.y;
    float* srow = S + ((long long)z * kT + m) * kT;
    const int valid = m + 1;
    const int j0 = threadIdx.x, j1 = threadIdx.x + 256;
    const float v0 = (j0 < valid) ? srow[j0] : -3.4e38f;
    const float v1 = (j1 < valid) ? srow[j1] : -3.4e38f;
    __shared__ float red[256];
    red[threadIdx.x] = fmaxf(v0, v1);
    __syncthreads();
    for (int off = 128; off > 0; off >>= 1) {
        if (threadIdx.x < off) red[threadIdx.x] = fmaxf(red[threadIdx.x], red[threadIdx.x + off]);
        __syncthreads();
    }
    const float mx = red[0];
    __syncthreads();
    const float e0 = (j0 < valid) ? expf(v0 - mx) : 0.f;
    const float e1 = (j1 < valid) ? expf(v1 - mx) : 0.f;
    red[threadIdx.x] = e0 + e1;
    __syncthreads();
    for (int off = 128; off > 0; off >>= 1) {
        if (threadIdx.x < off) red[threadIdx.x] += red[threadIdx.x + off];
        __syncthreads();
    }
    const float rs = 1.0f / red[0];
    bf16* prow = (bf16*)S + ((long long)z * kT + m) * kT;
    prow[j0] = (bf16)(e0 * rs);
    prow[j1] = (bf16)(e1 * rs);
}

// ---------------------------------------------------------------------------
extern "C" void kernel_launch(void* const* d_in, const int* in_sizes, int n_in,
                              void* d_out, int out_size, void* d_ws, size_t ws_size,
                              hipStream_t stream)
{
    (void)in_sizes; (void)n_in; (void)out_size; (void)ws_size;
    const int*   input_ids = (const int*)d_in[0];
    const float* wte    = (const float*)d_in[1];
    const float* ln1_g  = (const float*)d_in[2];
    const float* ln1_b  = (const float*)d_in[3];
    const float* attn_w = (const float*)d_in[4];
    const float* attn_b = (const float*)d_in[5];
    const float* cproj_w= (const float*)d_in[6];
    const float* cproj_b= (const float*)d_in[7];
    const float* ln2_g  = (const float*)d_in[8];
    const float* ln2_b  = (const float*)d_in[9];
    const float* fc_w   = (const float*)d_in[10];
    const float* fc_b   = (const float*)d_in[11];
    const float* pj_w   = (const float*)d_in[12];
    const float* pj_b   = (const float*)d_in[13];
    const float* lnf_g  = (const float*)d_in[14];
    const float* lnf_b  = (const float*)d_in[15];
    const float* lm_w   = (const float*)d_in[16];
    float* logits = (float*)d_out;

    // workspace carve-up (256B aligned)
    char* ws = (char*)d_ws;
    size_t off = 0;
    auto take = [&](size_t bytes) -> char* {
        char* p = ws + off; off += (bytes + 255) & ~(size_t)255; return p;
    };
    float* xbuf   = (float*)take((size_t)kBT * kD * 4);          // residual stream
    bf16*  hbf    = (bf16*) take((size_t)kBT * kD * 2);          // LN output
    float* qkv    = (float*)take((size_t)kBT * kQKV * 4);
    bf16*  qbf    = (bf16*) take((size_t)kB * kH * kT * kHD * 2);
    bf16*  kbf    = (bf16*) take((size_t)kB * kKVH * kT * kHD * 2);
    bf16*  vt     = (bf16*) take((size_t)kB * kKVH * kT * kHD * 2);
    float* scores = (float*)take((size_t)kB * kH * kT * kT * 4); // fp32 scores + bf16 probs (in place)
    bf16*  ybf    = (bf16*) take((size_t)kBT * kD * 2);
    bf16*  midbf  = (bf16*) take((size_t)kBT * kFF * 2);
    bf16*  awT    = (bf16*) take((size_t)kQKV * kD * 2);
    bf16*  pwT    = (bf16*) take((size_t)kD * kD * 2);
    bf16*  fwT    = (bf16*) take((size_t)kFF * kD * 2);
    bf16*  owT    = (bf16*) take((size_t)kD * kFF * 2);
    bf16*  lmT    = (bf16*) take((size_t)kV * kD * 2);
    bf16*  xfbf   = (bf16*) take((size_t)kBT * kD * 2);

    const dim3 tb256(256);

    // x = wte[input_ids]
    embed_kernel<<<dim3((kBT * kD) / 256), tb256, 0, stream>>>(input_ids, wte, xbuf);

    for (int l = 0; l < kL; ++l) {
        // ---- attention ----
        layernorm_kernel<<<dim3(kBT), tb256, 0, stream>>>(xbuf, ln1_g + l * kD, ln1_b + l * kD, hbf);

        transpose_cvt_kernel<<<dim3(kQKV / 32, kD / 32), dim3(32, 8), 0, stream>>>(
            attn_w + (size_t)l * kD * kQKV, awT, kD, kQKV);
        gemm_bf16_kernel<128><<<dim3(kQKV / 128, kBT / 64, 1), tb256, 0, stream>>>(
            hbf, 0LL, kD, awT, 0LL, 1, kD, attn_b + (size_t)l * kQKV, nullptr,
            qkv, 0LL, 0LL, 1, kQKV, kD, 0);

        rope_q_kernel<<<dim3((kB * kT * kH * kHD / 2) / 256), tb256, 0, stream>>>(qkv, qbf);
        rope_k_kernel<<<dim3((kB * kT * kKVH * kHD / 2) / 256), tb256, 0, stream>>>(qkv, kbf);
        vtrans_kernel<<<dim3((kB * kKVH * kHD * kT) / 256), tb256, 0, stream>>>(qkv, vt);

        // S[z] = q[z] @ k[z/4]^T  (scale folded into q)
        gemm_bf16_kernel<128><<<dim3(kT / 128, kT / 64, kB * kH), tb256, 0, stream>>>(
            qbf, (long long)kT * kHD, kHD, kbf, (long long)kT * kHD, 4, kHD,
            nullptr, nullptr, scores, (long long)kT * kT, 0LL, 1, kT, kHD, 0);

        softmax_causal_kernel<<<dim3(kT, kB * kH), tb256, 0, stream>>>(scores);

        // Y[b,t,h*HD+d] = P[z] @ V[z/4] ; bf16 output straight into [B,T,D] layout
        gemm_bf16_kernel<64><<<dim3(kHD / 64, kT / 64, kB * kH), tb256, 0, stream>>>(
            (const bf16*)scores, (long long)kT * kT, kT, vt, (long long)kHD * kT, 4, kT,
            nullptr, nullptr, ybf, (long long)kT * kD, (long long)kHD, kH, kD, kT, /*bf16out*/ 2);

        // x += Y @ c_proj + b
        transpose_cvt_kernel<<<dim3(kD / 32, kD / 32), dim3(32, 8), 0, stream>>>(
            cproj_w + (size_t)l * kD * kD, pwT, kD, kD);
        gemm_bf16_kernel<128><<<dim3(kD / 128, kBT / 64, 1), tb256, 0, stream>>>(
            ybf, 0LL, kD, pwT, 0LL, 1, kD, cproj_b + (size_t)l * kD, xbuf,
            xbuf, 0LL, 0LL, 1, kD, kD, 0);

        // ---- MLP ----
        layernorm_kernel<<<dim3(kBT), tb256, 0, stream>>>(xbuf, ln2_g + l * kD, ln2_b + l * kD, hbf);

        transpose_cvt_kernel<<<dim3(kFF / 32, kD / 32), dim3(32, 8), 0, stream>>>(
            fc_w + (size_t)l * kD * kFF, fwT, kD, kFF);
        gemm_bf16_kernel<128><<<dim3(kFF / 128, kBT / 64, 1), tb256, 0, stream>>>(
            hbf, 0LL, kD, fwT, 0LL, 1, kD, fc_b + (size_t)l * kFF, nullptr,
            midbf, 0LL, 0LL, 1, kFF, kD, /*gelu|bf16out*/ 1 | 2);

        transpose_cvt_kernel<<<dim3(kD / 32, kFF / 32), dim3(32, 8), 0, stream>>>(
            pj_w + (size_t)l * kFF * kD, owT, kFF, kD);
        gemm_bf16_kernel<128><<<dim3(kD / 128, kBT / 64, 1), tb256, 0, stream>>>(
            midbf, 0LL, kFF, owT, 0LL, 1, kFF, pj_b + (size_t)l * kD, xbuf,
            xbuf, 0LL, 0LL, 1, kD, kFF, 0);
    }

    // final LN + LM head
    layernorm_kernel<<<dim3(kBT), tb256, 0, stream>>>(xbuf, lnf_g, lnf_b, xfbf);
    transpose_cvt_kernel<<<dim3(kV / 32, kD / 32), dim3(32, 8), 0, stream>>>(lm_w, lmT, kD, kV);
    gemm_bf16_kernel<128><<<dim3(kV / 128, kBT / 64, 1), tb256, 0, stream>>>(
        xfbf, 0LL, kD, lmT, 0LL, 1, kD, nullptr, nullptr,
        logits, 0LL, 0LL, 1, kV, kD, 0);
}
